// PairwiseFlowEncoder_7687991460604
// MI455X (gfx1250) — matrix operand order repaired and checked by reading, not compiled, not run on hardware
//
#include <hip/hip_runtime.h>

typedef __attribute__((ext_vector_type(16))) _Float16 v16h;
typedef __attribute__((ext_vector_type(8)))  _Float16 v8h;
typedef __attribute__((ext_vector_type(8)))  float    v8f;

static __device__ __forceinline__ v16h zero_v16h() {
    v16h z;
#pragma unroll
    for (int i = 0; i < 16; ++i) z[i] = (_Float16)0.f;
    return z;
}

// ---------------------------------------------------------------------------
// Weight prep (layer 0 only): im2col layout [CoPad][Kpad], BN folded, f16.
// ---------------------------------------------------------------------------
__global__ __launch_bounds__(256)
void prep_weights_im2col_kernel(const float* __restrict__ w, const float* __restrict__ g,
                                const float* __restrict__ bsrc, const float* __restrict__ m,
                                const float* __restrict__ var,
                                _Float16* __restrict__ wh, float* __restrict__ bias,
                                int Co, int CoPad, int Kreal, int Kpad)
{
    int idx = blockIdx.x * blockDim.x + threadIdx.x;
    if (idx >= CoPad * Kpad) return;
    int co = idx / Kpad;
    int k  = idx - co * Kpad;
    float s = 1.f, val = 0.f;
    if (co < Co) {
        if (g) s = g[co] * rsqrtf(var[co] + 1e-5f);
        if (k < Kreal) val = w[co * Kreal + k] * s;
    }
    wh[idx] = (_Float16)val;
    if (k == 0) bias[co] = (co < Co) ? (g ? (bsrc[co] - m[co] * s) : bsrc[co]) : 0.f;
}

// ---------------------------------------------------------------------------
// Weight prep (NHWC layers): repack [Co][Ci][3][3] -> [tap][CoP][CiP] f16 with
// BN scale folded and zero padding. fuse_perm remaps fuse0's input channels to
// our fused layout [f1(0..63) | f2(64..127) | corr(128..208) | pad(209..223)]
// from the reference order [f1 | corr | f2].
// ---------------------------------------------------------------------------
__global__ __launch_bounds__(256)
void prep_weights_nhwc_kernel(const float* __restrict__ w, const float* __restrict__ g,
                              const float* __restrict__ bsrc, const float* __restrict__ m,
                              const float* __restrict__ var,
                              _Float16* __restrict__ wh, float* __restrict__ bias,
                              int Co, int CoP, int Ci, int CiP, int fuse_perm)
{
    int idx = blockIdx.x * blockDim.x + threadIdx.x;
    int per_tap = CoP * CiP;
    if (idx >= 9 * per_tap) return;
    int tap = idx / per_tap;
    int rem = idx - tap * per_tap;
    int co  = rem / CiP;
    int ci  = rem - co * CiP;
    float s = 1.f, val = 0.f;
    if (co < Co) {
        if (g) s = g[co] * rsqrtf(var[co] + 1e-5f);
        int cis = ci;
        if (fuse_perm)
            cis = (ci < 64) ? ci : (ci < 128) ? ci + 81 : (ci < 209) ? ci - 64 : -1;
        if (cis >= 0 && cis < Ci)
            val = w[(co * Ci + cis) * 9 + tap] * s;
    }
    wh[idx] = (_Float16)val;
    if (tap == 0 && ci == 0)
        bias[co] = (co < Co) ? (g ? (bsrc[co] - m[co] * s) : bsrc[co]) : 0.f;
}

// ---------------------------------------------------------------------------
// Layer 0: 3x3 s2 conv, Ci=3 (K=27, one WMMA chunk) from f32 NCHW image,
// im2col gather, output NHWC f16 padded to 32 channels (pad written as 0).
// ---------------------------------------------------------------------------
__global__ __launch_bounds__(32)
void conv0_wmma_kernel(const float* __restrict__ img, const _Float16* __restrict__ wh,
                       const float* __restrict__ bias, _Float16* __restrict__ out)
{
    const int lane = threadIdx.x & 31;
    const int half = lane >> 4, lm = lane & 15;
    const int n  = (blockIdx.x << 4) + lm;     // pixel in 256x256
    const int mt = blockIdx.y;                 // 0..1 over 32 padded channels
    const int b  = blockIdx.z;

    const int ho = n >> 8, wo = n & 255;
    const int hbase = 2 * ho - 1, wbase = 2 * wo - 1;
    const float* __restrict__ xb = img + (long long)b * 3 * 512 * 512;

    // A fragment (16-bit A 16x32 layout)
    const int arow = (mt << 4) + lm;
    const _Float16* wrow = wh + arow * 32;
    v16h a;
#pragma unroll
    for (int v = 0; v < 8; ++v) {
        int k = (half << 4) + (v << 1);
        a[2 * v]     = wrow[k];
        a[2 * v + 1] = wrow[k + 1];
    }
    // B fragment: bm[i] holds k = 16*half + i
    v16h bm = zero_v16h();
#pragma unroll
    for (int e = 0; e < 16; ++e) {
        int k = (half << 4) + e;
        if (k < 27) {
            int ci = k / 9, r = k - ci * 9;
            int kh = r / 3, kw = r - kh * 3;
            int hi = hbase + kh, wi = wbase + kw;
            float val = 0.f;
            if ((unsigned)hi < 512u && (unsigned)wi < 512u)
                val = xb[((long long)ci * 512 + hi) * 512 + wi];
            bm[e] = (_Float16)val;
        }
    }
    v8f acc = {0.f, 0.f, 0.f, 0.f, 0.f, 0.f, 0.f, 0.f};
    acc = __builtin_amdgcn_wmma_f32_16x16x32_f16(false, a, false, bm,
                                                 (short)0, acc, false, false);

    const int mbase = (mt << 4) + (half << 3);
    v8h oh;
#pragma unroll
    for (int v = 0; v < 8; ++v) {
        float y = acc[v] + bias[mbase + v];
        y = (y > 0.f) ? y : 0.1f * y;
        oh[v] = (_Float16)y;
    }
    *(v8h*)(out + ((long long)b * 65536 + n) * 32 + mbase) = oh;
}

// ---------------------------------------------------------------------------
// Generic 3x3 conv on NHWC f16 activations: 9 shifted 1x1 GEMMs on
// v_wmma_f32_16x16x32_f16. One wave per block computes 16 Co x 32 pixels
// (two N tiles share each A fragment). CIP (padded Ci) is a template arg so
// the tap x K-chunk loops fully unroll: all loads are contiguous b128s.
// Epilogue: bias + LeakyReLU, writes NHWC f16 (padded, pad = exact 0) and/or
// NCHW f32 (for d_out taps).
// ---------------------------------------------------------------------------
template <int CIP>
__global__ __launch_bounds__(32)
void conv_nhwc_wmma_kernel(const _Float16* __restrict__ x, const _Float16* __restrict__ wh,
                           const float* __restrict__ bias,
                           _Float16* __restrict__ onhwc, int ocp, int choff,
                           float* __restrict__ onchw, long long bstride, int Co,
                           int Hin, int Wout, int wshift, int stride,
                           int HWout, int CoP, int leaky)
{
    const int lane = threadIdx.x & 31;
    const int half = lane >> 4, lm = lane & 15;
    const int n0 = blockIdx.x << 5;
    const int mt = blockIdx.y;
    const int b  = blockIdx.z;
    const int Win = Hin;

    const _Float16* __restrict__ xb = x + (long long)b * Hin * Win * CIP;

    const int p0 = n0 + lm, p1 = p0 + 16;
    const int ho0 = p0 >> wshift, wo0 = p0 & (Wout - 1);
    const int ho1 = p1 >> wshift, wo1 = p1 & (Wout - 1);
    const int arow = (mt << 4) + lm;

    v8f acc0 = {0.f, 0.f, 0.f, 0.f, 0.f, 0.f, 0.f, 0.f};
    v8f acc1 = {0.f, 0.f, 0.f, 0.f, 0.f, 0.f, 0.f, 0.f};
    const v16h vz = zero_v16h();

#pragma unroll
    for (int kh = 0; kh < 3; ++kh) {
#pragma unroll
        for (int kw = 0; kw < 3; ++kw) {
            const int tap = kh * 3 + kw;
            const int hi0 = ho0 * stride + kh - 1, wi0 = wo0 * stride + kw - 1;
            const int hi1 = ho1 * stride + kh - 1, wi1 = wo1 * stride + kw - 1;
            const bool ok0 = (unsigned)hi0 < (unsigned)Hin && (unsigned)wi0 < (unsigned)Win;
            const bool ok1 = (unsigned)hi1 < (unsigned)Hin && (unsigned)wi1 < (unsigned)Win;
            const _Float16* __restrict__ xp0 = xb + ((long long)hi0 * Win + wi0) * CIP + (half << 4);
            const _Float16* __restrict__ xp1 = xb + ((long long)hi1 * Win + wi1) * CIP + (half << 4);
            const _Float16* __restrict__ wr  = wh + ((long long)tap * CoP + arow) * CIP + (half << 4);
#pragma unroll
            for (int kc = 0; kc < CIP; kc += 32) {
                // A: one contiguous 16-half run (a[i] = k 16*half+i)
                v8h alo = *(const v8h*)(wr + kc);
                v8h ahi = *(const v8h*)(wr + kc + 8);
                v16h a = __builtin_shufflevector(alo, ahi,
                        0, 1, 2, 3, 4, 5, 6, 7, 8, 9, 10, 11, 12, 13, 14, 15);
                // B: one contiguous 32B channel run per lane (b[i] = k 16*half+i)
                v16h b0 = ok0 ? *(const v16h*)(xp0 + kc) : vz;
                v16h b1 = ok1 ? *(const v16h*)(xp1 + kc) : vz;
                acc0 = __builtin_amdgcn_wmma_f32_16x16x32_f16(false, a, false, b0,
                                                              (short)0, acc0, false, false);
                acc1 = __builtin_amdgcn_wmma_f32_16x16x32_f16(false, a, false, b1,
                                                              (short)0, acc1, false, false);
            }
        }
    }

    const int mbase = (mt << 4) + (half << 3);
#pragma unroll
    for (int tile = 0; tile < 2; ++tile) {
        const v8f acc = tile ? acc1 : acc0;
        const int p = n0 + (tile << 4) + lm;
        float y[8];
#pragma unroll
        for (int v = 0; v < 8; ++v) {
            float t = acc[v] + bias[mbase + v];
            if (leaky) t = (t > 0.f) ? t : 0.1f * t;
            y[v] = t;
        }
        if (onhwc) {
            v8h oh;
#pragma unroll
            for (int v = 0; v < 8; ++v) oh[v] = (_Float16)y[v];
            *(v8h*)(onhwc + ((long long)b * HWout + p) * ocp + choff + mbase) = oh;
        }
        if (onchw) {
#pragma unroll
            for (int v = 0; v < 8; ++v) {
                int mo = mbase + v;
                if (mo < Co)
                    onchw[(long long)b * bstride + (long long)mo * HWout + p] = y[v];
            }
        }
    }
}

// ---------------------------------------------------------------------------
// L2-normalize 64 channels read contiguously from the f16 fused buffer
// (stride 224), write f32 NHWC for the correlation.
// ---------------------------------------------------------------------------
__global__ __launch_bounds__(256)
void l2norm_nhwc_kernel(const _Float16* __restrict__ fused, float* __restrict__ out,
                        int choff, int total)
{
    int idx = blockIdx.x * blockDim.x + threadIdx.x;
    if (idx >= total) return;
    const _Float16* fp = fused + (long long)idx * 224 + choff;
    float vals[64];
    float ss = 0.f;
#pragma unroll
    for (int c = 0; c < 64; ++c) { vals[c] = (float)fp[c]; ss += vals[c] * vals[c]; }
    float inv = 1.f / fmaxf(sqrtf(ss), 1e-12f);
    float* op = out + (long long)idx * 64;
#pragma unroll
    for (int c = 0; c < 64; ++c) op[c] = vals[c] * inv;
}

// ---------------------------------------------------------------------------
// PWC local correlation, radius 4 -> 81 channels. f1 row cached in registers,
// contiguous NHWC f32 loads; writes f32 NCHW to d_out and f16 into fused
// channels 128..208.
// ---------------------------------------------------------------------------
__global__ __launch_bounds__(64)
void corr_kernel(const float* __restrict__ f1n, const float* __restrict__ f2n,
                 float* __restrict__ out1, _Float16* __restrict__ fused, int total)
{
    int idx = blockIdx.x * blockDim.x + threadIdx.x;
    if (idx >= total) return;
    int b = idx >> 12;
    int p = idx & 4095;
    int h = p >> 6, w = p & 63;

    const float* a = f1n + (long long)idx * 64;
    float areg[64];
#pragma unroll
    for (int c = 0; c < 64; ++c) areg[c] = a[c];
    _Float16* fp = fused + (long long)idx * 224 + 128;

#pragma unroll 1
    for (int dy = -4; dy <= 4; ++dy) {
        int hh = h + dy;
#pragma unroll 1
        for (int dx = -4; dx <= 4; ++dx) {
            int ww = w + dx;
            float s = 0.f;
            if ((unsigned)hh < 64u && (unsigned)ww < 64u) {
                const float* bb = f2n + (((long long)b << 12) + (hh << 6) + ww) * 64;
#pragma unroll
                for (int c = 0; c < 64; ++c) s += areg[c] * bb[c];
            }
            int d = (dy + 4) * 9 + (dx + 4);
            out1[((long long)b * 81 + d) * 4096 + p] = s;
            fp[d] = (_Float16)s;
        }
    }
}

// ---------------------------------------------------------------------------
extern "C" void kernel_launch(void* const* d_in, const int* in_sizes, int n_in,
                              void* d_out, int out_size, void* d_ws, size_t ws_size,
                              hipStream_t stream)
{
    (void)in_sizes; (void)n_in; (void)out_size; (void)ws_size;
    const float* img1 = (const float*)d_in[0];
    const float* img2 = (const float*)d_in[1];

    // sorted-key pytree flatten: feat(6 blks), flow(blk + {b,w}), fuse(2 blks);
    // each blk flattens as b,g,m,v,w.
    struct LayerP { const float *w, *g, *b, *m, *v; };
    LayerP lp[10];
    for (int i = 0; i < 6; ++i) {
        int base = 2 + 5 * i;
        lp[i] = { (const float*)d_in[base + 4], (const float*)d_in[base + 1],
                  (const float*)d_in[base + 0], (const float*)d_in[base + 2],
                  (const float*)d_in[base + 3] };
    }
    lp[6] = { (const float*)d_in[43], (const float*)d_in[40], (const float*)d_in[39],
              (const float*)d_in[41], (const float*)d_in[42] };                 // fuse[0]
    lp[7] = { (const float*)d_in[48], (const float*)d_in[45], (const float*)d_in[44],
              (const float*)d_in[46], (const float*)d_in[47] };                 // fuse[1]
    lp[8] = { (const float*)d_in[36], (const float*)d_in[33], (const float*)d_in[32],
              (const float*)d_in[34], (const float*)d_in[35] };                 // flow[0]
    lp[9] = { (const float*)d_in[38], nullptr, (const float*)d_in[37],
              nullptr, nullptr };                                               // flow[1]

    // NHWC layers 1..9 (index into these arrays is layer-1)
    static const int Ci_[9]  = {16, 16, 32, 32, 64, 209, 128, 128, 64};
    static const int CIP_[9] = {32, 32, 32, 32, 64, 224, 128, 128, 64};
    static const int Co_[9]  = {16, 32, 32, 64, 64, 128, 128, 64, 2};
    static const int CoP_[9] = {32, 32, 32, 64, 64, 128, 128, 64, 16};
    static const int Hin_[9] = {256, 256, 128, 128, 64, 64, 64, 64, 64};
    static const int Str_[9] = {1, 2, 1, 2, 1, 1, 1, 1, 1};

    // ---- workspace carve-up ----
    char* ws = (char*)d_ws;
    size_t off = 0;
    auto take = [&](size_t bytes) -> char* {
        char* p = ws + off;
        off = (off + bytes + 255) & ~(size_t)255;
        return p;
    };

    _Float16* wh0 = (_Float16*)take(32 * 32 * sizeof(_Float16));
    float*    bs0 = (float*)take(32 * sizeof(float));
    _Float16* wh[9]; float* bias[9];
    for (int i = 0; i < 9; ++i) {
        wh[i]   = (_Float16*)take((size_t)9 * CoP_[i] * CIP_[i] * sizeof(_Float16));
        bias[i] = (float*)take((size_t)CoP_[i] * sizeof(float));
    }
    _Float16* nb0    = (_Float16*)take((size_t)8 * 256 * 256 * 32 * sizeof(_Float16));
    _Float16* nb1    = (_Float16*)take((size_t)8 * 256 * 256 * 32 * sizeof(_Float16));
    _Float16* fusedh = (_Float16*)take((size_t)8 * 4096 * 224 * sizeof(_Float16));
    float* f1n = (float*)nb0;                         // reused between L5 and fuse0
    float* f2n = f1n + (size_t)8 * 4096 * 64;

    float* out    = (float*)d_out;
    float* o_f1   = out;                 // [8,64,64,64]
    float* o_f2   = out + 2097152;       // [8,64,64,64]
    float* o_corr = out + 4194304;       // [8,81,64,64]
    float* o_pair = out + 6848512;       // [8,128,64,64]
    float* o_flow = out + 11042816;      // [8,2,64,64]

    // ---- weight prep ----
    prep_weights_im2col_kernel<<<(32 * 32 + 255) / 256, 256, 0, stream>>>(
        lp[0].w, lp[0].g, lp[0].b, lp[0].m, lp[0].v, wh0, bs0, 16, 32, 27, 32);
    for (int i = 0; i < 9; ++i) {
        int total = 9 * CoP_[i] * CIP_[i];
        prep_weights_nhwc_kernel<<<(total + 255) / 256, 256, 0, stream>>>(
            lp[i + 1].w, lp[i + 1].g, lp[i + 1].b, lp[i + 1].m, lp[i + 1].v,
            wh[i], bias[i], Co_[i], CoP_[i], Ci_[i], CIP_[i], (i == 5) ? 1 : 0);
    }

    // ---- conv launcher (CIP template dispatch) ----
    auto conv = [&](int i, const _Float16* x, _Float16* onhwc, int ocp, int choff,
                    float* onchw, long long bstride, int leaky) {
        int Wout = Hin_[i] / Str_[i];
        int wsh = 0; while ((1 << wsh) < Wout) ++wsh;
        int HWout = Wout * Wout;
        dim3 grid(HWout / 32, CoP_[i] / 16, 8);
        dim3 blk(32);
#define CONV_CASE(CIPV)                                                          \
        conv_nhwc_wmma_kernel<CIPV><<<grid, blk, 0, stream>>>(                   \
            x, wh[i], bias[i], onhwc, ocp, choff, onchw, bstride, Co_[i],        \
            Hin_[i], Wout, wsh, Str_[i], HWout, CoP_[i], leaky)
        switch (CIP_[i]) {
        case 32:  CONV_CASE(32);  break;
        case 64:  CONV_CASE(64);  break;
        case 128: CONV_CASE(128); break;
        case 224: CONV_CASE(224); break;
        }
#undef CONV_CASE
    };

    // ---- backbone, image 1 -> fused ch 0..63 + d_out f1 ----
    conv0_wmma_kernel<<<dim3(4096, 2, 8), dim3(32), 0, stream>>>(img1, wh0, bs0, nb0);
    conv(0, nb0, nb1, 32, 0, nullptr, 0, 1);
    conv(1, nb1, nb0, 32, 0, nullptr, 0, 1);
    conv(2, nb0, nb1, 32, 0, nullptr, 0, 1);
    conv(3, nb1, nb0, 64, 0, nullptr, 0, 1);
    conv(4, nb0, fusedh, 224, 0, o_f1, 64ll * 4096, 1);

    // ---- backbone, image 2 -> fused ch 64..127 + d_out f2 ----
    conv0_wmma_kernel<<<dim3(4096, 2, 8), dim3(32), 0, stream>>>(img2, wh0, bs0, nb0);
    conv(0, nb0, nb1, 32, 0, nullptr, 0, 1);
    conv(1, nb1, nb0, 32, 0, nullptr, 0, 1);
    conv(2, nb0, nb1, 32, 0, nullptr, 0, 1);
    conv(3, nb1, nb0, 64, 0, nullptr, 0, 1);
    conv(4, nb0, fusedh, 224, 64, o_f2, 64ll * 4096, 1);

    // ---- l2norm + correlation -> fused ch 128..208 + d_out corr ----
    int tot = 8 * 4096;
    l2norm_nhwc_kernel<<<(tot + 255) / 256, 256, 0, stream>>>(fusedh, f1n, 0, tot);
    l2norm_nhwc_kernel<<<(tot + 255) / 256, 256, 0, stream>>>(fusedh, f2n, 64, tot);
    corr_kernel<<<(tot + 63) / 64, 64, 0, stream>>>(f1n, f2n, o_corr, fusedh, tot);

    // ---- fuse + flow head ----
    conv(5, fusedh, nb0, 128, 0, nullptr, 0, 1);                 // fuse0 (perm'd weights)
    conv(6, nb0, nb1, 128, 0, o_pair, 128ll * 4096, 1);          // fuse1 -> pair_feat
    conv(7, nb1, nb0, 64, 0, nullptr, 0, 1);                     // flow0
    conv(8, nb0, nullptr, 0, 0, o_flow, 2ll * 4096, 0);          // flow1 (bias, no BN/leaky)
}
